// FeaturePropagation_51024211476519
// MI455X (gfx1250) — matrix operand order repaired and checked by reading, not compiled, not run on hardware
//
#include <hip/hip_runtime.h>
#include <hip/hip_bf16.h>
#include <stdint.h>

// ---------------------------------------------------------------------------
// FeaturePropagation on MI455X (gfx1250):
//   10x { h = A @ h (COO SpMM); h[known] = x[known] }
//
// Each workgroup owns a 16-row dst block; per 32-edge group it computes
//   C(16x256, f32 VGPRs) += A(16x32 bf16 weight-selection) @ B(32x16 bf16)
// with v_wmma_f32_16x16x32_bf16.  Gathered source rows are staged to LDS in
// a ROW-PAIR INTERLEAVED layout so every B-fragment VGPR is one aligned
// ds_load_b32 (no u16 loads / half-reg merges).  Edge meta is broadcast to
// LDS via global_load_async_to_lds_b32 (ASYNCcnt path).
// h lives in bf16 (25.6MB, fully L2-resident on the 192MB L2).
// ---------------------------------------------------------------------------

#define N_GENES   256
#define BLK_ROWS  16
#define GRP_EDGES 32
#define NUM_ITERS 10
// interleaved tile: (k,g) -> (k>>1)*PAIR_STRIDE_B + g*4 + (k&1)*2
#define PAIR_STRIDE_B 1056          // 256 genes * 4B + 32B pad (264 u32)
#define PAIR_STRIDE_W 264           // in u32

typedef __attribute__((ext_vector_type(16))) __bf16 v16bf;
typedef __attribute__((ext_vector_type(8)))  float  v8f;

__device__ __forceinline__ unsigned short f2bf(float f) {
  unsigned u = __float_as_uint(f);
  unsigned r = u + 0x7FFFu + ((u >> 16) & 1u);   // round-to-nearest-even
  return (unsigned short)(r >> 16);
}

// ---------------- preprocessing: bin edges by dst block ----------------

__global__ void zero_i32_kernel(int* p, int n) {
  int i = blockIdx.x * blockDim.x + threadIdx.x;
  if (i < n) p[i] = 0;
}

__global__ void count_kernel(const long long* __restrict__ ei, int E,
                             int* __restrict__ counts) {
  int e = blockIdx.x * blockDim.x + threadIdx.x;
  if (e >= E) return;
  int dst = (int)ei[e];                 // edge_index[0] = dst
  atomicAdd(&counts[dst >> 4], 1);
}

__global__ void scan_kernel(const int* __restrict__ counts,
                            int* __restrict__ offs, int* __restrict__ cur,
                            int nb) {
  if (threadIdx.x == 0 && blockIdx.x == 0) {
    int acc = 0;
    for (int i = 0; i < nb; ++i) { offs[i] = acc; cur[i] = acc; acc += counts[i]; }
    offs[nb] = acc;
  }
}

__global__ void bin_kernel(const long long* __restrict__ ei,
                           const float* __restrict__ ew, int E,
                           int* __restrict__ cur,
                           int* __restrict__ bsrc,
                           unsigned* __restrict__ meta) {
  int e = blockIdx.x * blockDim.x + threadIdx.x;
  if (e >= E) return;
  int dst = (int)ei[e];
  int src = (int)ei[(size_t)E + e];     // edge_index[1] = src
  int pos = atomicAdd(&cur[dst >> 4], 1);
  bsrc[pos] = src;
  meta[pos] = ((unsigned)(dst & 15) << 16) | (unsigned)f2bf(ew[e]);
}

__global__ void cvt_kernel(const float* __restrict__ x,
                           unsigned short* __restrict__ hb, int n) {
  int i = blockIdx.x * blockDim.x + threadIdx.x;
  if (i < n) hb[i] = f2bf(x[i]);
}

// ---------------- main propagation step (one dst-block per WG) -------------

__launch_bounds__(256, 2)
__global__ void prop_kernel(const unsigned short* __restrict__ hin,   // bf16 bits
                            const float* __restrict__ orig,           // x (f32)
                            unsigned short* __restrict__ outb,        // bf16 out
                            float* __restrict__ outf,                 // f32 out (last iter)
                            const int* __restrict__ offs,
                            const int* __restrict__ bsrc,
                            const unsigned* __restrict__ meta,
                            int lastIter) {
  __shared__ unsigned tileW[16 * PAIR_STRIDE_W];   // 16 row-pairs, 16.5 KB
  __shared__ unsigned metaL[GRP_EDGES];            // group edge meta (async-staged)

  const int blk   = blockIdx.x;                 // dst rows [blk*16, blk*16+16)
  const int tid   = threadIdx.x;
  const int lane  = tid & 31;
  const int wave  = tid >> 5;                   // 8 waves; wave w -> gene tiles 2w,2w+1
  const int half  = lane >> 4;                  // 0: lanes 0-15, 1: lanes 16-31
  const int lm    = lane & 15;                  // A row (M) / B,C column (N)

  const int gstart_blk = offs[blk];
  const int gend_blk   = offs[blk + 1];

  v8f c0 = {}; v8f c1 = {};

  // staging role: row-pair rp = tid&15, gene segment gseg = tid>>4 (16 genes)
  const int rp   = tid & 15;
  const int gseg = tid >> 4;

  for (int gs = gstart_blk; gs < gend_blk; gs += GRP_EDGES) {
    __syncthreads();   // previous tile fully consumed before overwrite

    // ---- edge meta -> LDS via CDNA5 async-to-LDS (wave 0, 4B/lane) ----
    if (tid < GRP_EDGES) {
      int idx = gs + tid;
      if (idx >= gend_blk) idx = gend_blk - 1;          // clamp; guarded on read
      unsigned long long ga = (unsigned long long)(meta + idx);
      unsigned lds = (unsigned)(uintptr_t)(&metaL[tid]);
      asm volatile("global_load_async_to_lds_b32 %0, %1, off"
                   :: "v"(lds), "v"(ga) : "memory");
    }

    // ---- gather two source rows, zip into row-pair interleaved LDS ----
    // pair rp holds k=2rp (lo16) and k=2rp+1 (hi16), 16 genes per thread.
    {
      int eA = gs + 2 * rp, eB = eA + 1;
      int srcA = (eA < gend_blk) ? bsrc[eA] : 0;
      int srcB = (eB < gend_blk) ? bsrc[eB] : 0;
      const uint4* pA = (const uint4*)(hin + (size_t)srcA * N_GENES + gseg * 16);
      const uint4* pB = (const uint4*)(hin + (size_t)srcB * N_GENES + gseg * 16);
      uint4 ra0 = pA[0], ra1 = pA[1];                   // 16 genes of row 2rp
      uint4 rb0 = pB[0], rb1 = pB[1];                   // 16 genes of row 2rp+1
      union { uint4 v[2]; unsigned short s[16]; } a, b;
      a.v[0] = ra0; a.v[1] = ra1;
      b.v[0] = rb0; b.v[1] = rb1;
      uint4 o[4];
      unsigned* ow = (unsigned*)o;
#pragma unroll
      for (int g = 0; g < 16; ++g)
        ow[g] = (unsigned)a.s[g] | ((unsigned)b.s[g] << 16);
      uint4* dst = (uint4*)&tileW[rp * PAIR_STRIDE_W + gseg * 16];
#pragma unroll
      for (int i = 0; i < 4; ++i) dst[i] = o[i];        // 4x ds_store_b128
    }

    asm volatile("s_wait_asynccnt 0" ::: "memory");
    __syncthreads();   // zipped tile + meta visible to all waves

    // ---- A fragment: 16x32 bf16 weight-selection (from LDS meta) ----
    // 16-bit A 16x32 layout: lanes0-15: V0-3 K=0..7, V4-7 K=16..23;
    //                        lanes16-31: +8 on each K.
    union { v16bf v; unsigned short u[16]; } A;
#pragma unroll
    for (int j = 0; j < 8; ++j) {
      int k0 = ((j < 4) ? (2 * j) : (16 + 2 * (j - 4))) + (half ? 8 : 0);
#pragma unroll
      for (int s = 0; s < 2; ++s) {
        int k = k0 + s;
        unsigned m = metaL[k];                          // ds broadcast read
        bool valid = (gs + k) < gend_blk;
        A.u[2 * j + s] =
            (valid && (((m >> 16) & 15u) == (unsigned)lm))
                ? (unsigned short)(m & 0xFFFFu) : (unsigned short)0;
      }
    }

    // ---- B fragments: 8x ds_load_b32 each, then WMMA ----
    // B layout: lanes0-15 K=0..7,16..23 (pairs 0-3,8-11), lanes16-31 +8 on each K;
    // VGPR j = {k=2p+1, k=2p} with p = 4*half + j + (j>=4 ? 4 : 0); lane%16 = N.
    const int pbase = 4 * half;
#pragma unroll
    for (int t = 0; t < 2; ++t) {
      int col = (2 * wave + t) * 16 + lm;
      union { v16bf v; unsigned w[8]; } B;
#pragma unroll
      for (int j = 0; j < 8; ++j)
        B.w[j] = tileW[(pbase + j + ((j >= 4) ? 4 : 0)) * PAIR_STRIDE_W + col];
      if (t == 0)
        c0 = __builtin_amdgcn_wmma_f32_16x16x32_bf16(false, A.v, false, B.v,
                                                     (short)0, c0, false, false);
      else
        c1 = __builtin_amdgcn_wmma_f32_16x16x32_bf16(false, A.v, false, B.v,
                                                     (short)0, c1, false, false);
    }
  }

  // ---- epilogue: clamp known entries, store (bf16 mid-iters, f32 last) ----
  // C/D layout: VGPR v <-> M = v + 8*half; lane%16 = N.
#pragma unroll
  for (int t = 0; t < 2; ++t) {
    int gene = (2 * wave + t) * 16 + lm;
#pragma unroll
    for (int v = 0; v < 8; ++v) {
      int node = blk * BLK_ROWS + v + 8 * half;    // N=50000 = 3125*16, no OOB
      size_t idx = (size_t)node * N_GENES + gene;
      float o = orig[idx];
      float acc = (t == 0) ? c0[v] : c1[v];
      float val = (o != 0.0f) ? o : acc;
      if (lastIter) outf[idx] = val;
      else          outb[idx] = f2bf(val);
    }
  }
}

// ---------------------------------------------------------------------------

extern "C" void kernel_launch(void* const* d_in, const int* in_sizes, int n_in,
                              void* d_out, int out_size, void* d_ws, size_t ws_size,
                              hipStream_t stream) {
  const float*     x  = (const float*)d_in[0];
  const long long* ei = (const long long*)d_in[1];
  const float*     ew = (const float*)d_in[2];
  const int E  = in_sizes[2];
  const int NG = in_sizes[0];              // N_NODES * 256
  const int N  = NG / N_GENES;
  const int NB = (N + BLK_ROWS - 1) / BLK_ROWS;   // 3125 dst blocks

  auto align = [](size_t v) { return (v + 255) & ~(size_t)255; };
  uint8_t* w = (uint8_t*)d_ws;
  size_t o = 0;
  unsigned short* hA = (unsigned short*)(w + o); o = align(o + (size_t)NG * 2);
  unsigned short* hB = (unsigned short*)(w + o); o = align(o + (size_t)NG * 2);
  int*      counts = (int*)(w + o);       o = align(o + (size_t)NB * 4);
  int*      offs   = (int*)(w + o);       o = align(o + (size_t)(NB + 1) * 4);
  int*      cur    = (int*)(w + o);       o = align(o + (size_t)NB * 4);
  int*      bsrc   = (int*)(w + o);       o = align(o + (size_t)E * 4);
  unsigned* meta   = (unsigned*)(w + o);  o = align(o + (size_t)E * 4);

  // --- bin edges by dst block (edges are launch-invariant) ---
  zero_i32_kernel<<<(NB + 255) / 256, 256, 0, stream>>>(counts, NB);
  count_kernel<<<(E + 255) / 256, 256, 0, stream>>>(ei, E, counts);
  scan_kernel<<<1, 1, 0, stream>>>(counts, offs, cur, NB);
  bin_kernel<<<(E + 255) / 256, 256, 0, stream>>>(ei, ew, E, cur, bsrc, meta);

  // --- h0 = bf16(x) ---
  cvt_kernel<<<(NG + 255) / 256, 256, 0, stream>>>(x, hA, NG);

  // --- 10 propagation iterations; last writes f32 to d_out ---
  unsigned short* bufs[2] = {hA, hB};
  int c = 0;
  for (int it = 0; it < NUM_ITERS; ++it) {
    int last = (it == NUM_ITERS - 1);
    prop_kernel<<<NB, 256, 0, stream>>>(
        bufs[c], x,
        last ? (unsigned short*)nullptr : bufs[1 - c],
        last ? (float*)d_out : (float*)nullptr,
        offs, bsrc, meta, last);
    c ^= 1;
  }
}
